// AstraloraLayer_60748017435285
// MI455X (gfx1250) — matrix operand order repaired and checked by reading, not compiled, not run on hardware
//
#include <hip/hip_runtime.h>

// y[M,N] = x[M,K] @ w[N,K]^T  with M=65536, N=512, K=512, fp32 in/out.
// bf16 hi/lo split: acc += xh*wh + xh*wl + xl*wh via v_wmma_f32_16x16x32_bf16
// (fp32 accumulators) => ~fp32 accuracy at bf16 matrix-core rate.
// Block tile 128x128, K-step 32, double-buffered LDS (one barrier/iter),
// rows padded to 40 bf16 (80B) for conflict-free ds_load_b128.

typedef __bf16 v16bf __attribute__((ext_vector_type(16)));
typedef __bf16 v8bf  __attribute__((ext_vector_type(8)));
typedef __bf16 v4bf  __attribute__((ext_vector_type(4)));
typedef float  v8f   __attribute__((ext_vector_type(8)));

#define M_TOT 65536
#define N_TOT 512
#define K_TOT 512
#define BM 128
#define BN 128
#define BK 32
#define LDSW 40   // 32 K-elems + 8 pad (bf16) => 80B row stride

union Frag16 { v16bf v; v8bf h[2]; };

__global__ __launch_bounds__(256)
void astralora_gemm_bf16x3(const float* __restrict__ X,
                           const float* __restrict__ W,
                           float* __restrict__ Y) {
  __shared__ __bf16 sXh[2][BM * LDSW];
  __shared__ __bf16 sXl[2][BM * LDSW];
  __shared__ __bf16 sWh[2][BN * LDSW];
  __shared__ __bf16 sWl[2][BN * LDSW];

  const int tid   = threadIdx.x;
  const int lane  = tid & 31;
  const int wv    = tid >> 5;
  const int waveM = wv >> 2;   // 0..1
  const int waveN = wv & 3;    // 0..3
  const int mBase = blockIdx.y * BM;
  const int nBase = blockIdx.x * BN;

  float4 xr[4], wr[4];

  // ---- global fetch of one 128x32 x-tile and 128x32 w-tile into registers ----
  auto fetch = [&](int kt) {
    const int kBase = kt * BK;
    #pragma unroll
    for (int t = 0; t < 4; ++t) {
      const int idx = t * 256 + tid;
      const int row = idx >> 3;       // 8 float4 per 32-col row
      const int c4  = idx & 7;
      xr[t] = *(const float4*)(X + (size_t)(mBase + row) * K_TOT + kBase + c4 * 4);
      wr[t] = *(const float4*)(W + (size_t)(nBase + row) * K_TOT + kBase + c4 * 4);
    }
  };

  // ---- split fp32 -> (hi,lo) bf16 and stage into LDS buffer `buf` ----
  auto stage = [&](int buf) {
    #pragma unroll
    for (int t = 0; t < 4; ++t) {
      const int idx = t * 256 + tid;
      const int row = idx >> 3;
      const int c4  = idx & 7;
      const float xv[4] = {xr[t].x, xr[t].y, xr[t].z, xr[t].w};
      const float wvv[4] = {wr[t].x, wr[t].y, wr[t].z, wr[t].w};
      v4bf xh, xl, wh, wl;
      #pragma unroll
      for (int e = 0; e < 4; ++e) {
        __bf16 hx = (__bf16)xv[e];
        xh[e] = hx;
        xl[e] = (__bf16)(xv[e] - (float)hx);
        __bf16 hw = (__bf16)wvv[e];
        wh[e] = hw;
        wl[e] = (__bf16)(wvv[e] - (float)hw);
      }
      const int off = row * LDSW + c4 * 4;   // 8B-aligned
      *(v4bf*)&sXh[buf][off] = xh;
      *(v4bf*)&sXl[buf][off] = xl;
      *(v4bf*)&sWh[buf][off] = wh;
      *(v4bf*)&sWl[buf][off] = wl;
    }
  };

  v8f acc[4][2];
  #pragma unroll
  for (int i = 0; i < 4; ++i)
    #pragma unroll
    for (int j = 0; j < 2; ++j)
      #pragma unroll
      for (int e = 0; e < 8; ++e) acc[i][j][e] = 0.0f;

  const int rsel  = lane & 15;
  const int kloA  = (lane & 16) ? 8 : 0;    // A frag: K chunks {klo..+7, klo+16..+23}
  const int koffB = (lane & 16) ? 8 : 0;    // B frag: K chunks {klo..+7, klo+16..+23}

  // ---- compute all 24 WMMAs against LDS buffer `buf` ----
  auto compute = [&](int buf) {
    Frag16 ah[4], al[4], bh[2], bl[2];
    #pragma unroll
    for (int i = 0; i < 4; ++i) {
      const int row  = waveM * 64 + i * 16 + rsel;
      const int base = row * LDSW + kloA;            // 16B-aligned
      ah[i].h[0] = *(const v8bf*)&sXh[buf][base];
      ah[i].h[1] = *(const v8bf*)&sXh[buf][base + 16];
      al[i].h[0] = *(const v8bf*)&sXl[buf][base];
      al[i].h[1] = *(const v8bf*)&sXl[buf][base + 16];
    }
    #pragma unroll
    for (int j = 0; j < 2; ++j) {
      const int row  = waveN * 32 + j * 16 + rsel;
      const int base = row * LDSW + koffB;           // 16B-aligned
      bh[j].h[0] = *(const v8bf*)&sWh[buf][base];
      bh[j].h[1] = *(const v8bf*)&sWh[buf][base + 16];
      bl[j].h[0] = *(const v8bf*)&sWl[buf][base];
      bl[j].h[1] = *(const v8bf*)&sWl[buf][base + 16];
    }
    #pragma unroll
    for (int i = 0; i < 4; ++i)
      #pragma unroll
      for (int j = 0; j < 2; ++j) {
        acc[i][j] = __builtin_amdgcn_wmma_f32_16x16x32_bf16(
            false, ah[i].v, false, bh[j].v, (short)0, acc[i][j], false, false);
        acc[i][j] = __builtin_amdgcn_wmma_f32_16x16x32_bf16(
            false, ah[i].v, false, bl[j].v, (short)0, acc[i][j], false, false);
        acc[i][j] = __builtin_amdgcn_wmma_f32_16x16x32_bf16(
            false, al[i].v, false, bh[j].v, (short)0, acc[i][j], false, false);
      }
  };

  const int KT = K_TOT / BK;   // 16

  fetch(0);
  stage(0);
  __syncthreads();

  for (int kt = 0; kt < KT; ++kt) {
    const int cur = kt & 1;
    if (kt + 1 < KT) fetch(kt + 1);   // global loads in flight under compute
    compute(cur);                      // ds reads of `cur` + 24 WMMAs
    if (kt + 1 < KT) stage(cur ^ 1);   // LDS writes to other buffer (no race)
    __syncthreads();                   // single barrier per iteration
  }

  // ---- store C: VGPR v -> M = v (lanes 0-15) / v+8 (lanes 16-31), N = lane%16 ----
  #pragma unroll
  for (int i = 0; i < 4; ++i) {
    const int rbase = mBase + waveM * 64 + i * 16 + ((lane & 16) ? 8 : 0);
    #pragma unroll
    for (int j = 0; j < 2; ++j) {
      const int col = nBase + waveN * 32 + j * 16 + rsel;
      #pragma unroll
      for (int v = 0; v < 8; ++v)
        Y[(size_t)(rbase + v) * N_TOT + col] = acc[i][j][v];
    }
  }
}

extern "C" void kernel_launch(void* const* d_in, const int* in_sizes, int n_in,
                              void* d_out, int out_size, void* d_ws, size_t ws_size,
                              hipStream_t stream) {
  const float* x = (const float*)d_in[0];   // (8, 8192, 512) fp32
  const float* w = (const float*)d_in[1];   // (512*512,) fp32, row-major (DOUT, DIN)
  float* out = (float*)d_out;               // (8, 8192, 512) fp32
  (void)in_sizes; (void)n_in; (void)out_size; (void)d_ws; (void)ws_size;

  dim3 grid(N_TOT / BN, M_TOT / BM);        // (4, 512)
  astralora_gemm_bf16x3<<<grid, dim3(256), 0, stream>>>(x, w, out);
}
